// CARAFE_18485539242296
// MI455X (gfx1250) — compile-verified, matrix operand order checked by reading, not run on hardware
//
#include <hip/hip_runtime.h>

// ---------------------------------------------------------------------------
// CARAFE on MI455X (gfx1250, wave32).
// Dominant cost: conv_out 1x1 GEMM (512x512x131072 = 68.7 GFLOP) -> bf16 WMMA.
// conv_down 1x1 GEMM (128x512x32768 = 4.3 GFLOP) -> bf16 WMMA.
// 3x3 conv / softmax / reassembly are small (<4 GFLOP total) -> VALU fp32.
// Data path: x -> NHWC bf16 once; reassembly emits NHWC bf16 so both GEMM "B"
// operands stream K-contiguously (two b128 loads per lane per WMMA frag).
// ---------------------------------------------------------------------------

typedef __bf16 bf16;
typedef __attribute__((ext_vector_type(16))) bf16  v16bf;
typedef __attribute__((ext_vector_type(8)))  bf16  bf16x8;
typedef __attribute__((ext_vector_type(4)))  bf16  bf16x4;
typedef __attribute__((ext_vector_type(8)))  float v8f;

// Problem constants
#define NN    8
#define CIN   512
#define MIDC  128
#define ENCC  36
#define HH    64
#define WW    64
#define S1    4096      // H*W
#define H2    128
#define W2    128
#define S2    16384     // 2H*2W
#define STOT1 (NN * S1)     // 32768 columns for GEMM1
#define STOT2 (NN * S2)     // 131072 columns for GEMM2

__device__ __forceinline__ bf16 f2bf(float f) {
    unsigned u = __builtin_bit_cast(unsigned, f);
    u += 0x7FFFu + ((u >> 16) & 1u);                 // round-to-nearest-even
    unsigned short h = (unsigned short)(u >> 16);
    return __builtin_bit_cast(bf16, h);
}
__device__ __forceinline__ float bf2f(bf16 b) {
    unsigned short h = __builtin_bit_cast(unsigned short, b);
    unsigned u = ((unsigned)h) << 16;
    return __builtin_bit_cast(float, u);
}

// 16-bit A/B fragment for V_WMMA_*_16X16X32: lane covers row/col (lane%16),
// kb = (lane<16)?0:8; elements 0..7 -> K = k0+kb+e, elements 8..15 -> +16.
__device__ __forceinline__ v16bf load_frag16x32(const bf16* __restrict__ row,
                                                int k0, int kb) {
    bf16x8 lo = *(const bf16x8*)(row + k0 + kb);
    bf16x8 hi = *(const bf16x8*)(row + k0 + kb + 16);
    return __builtin_shufflevector(lo, hi, 0, 1, 2, 3, 4, 5, 6, 7,
                                           8, 9, 10, 11, 12, 13, 14, 15);
}

// ---------------------------------------------------------------- fp32->bf16
__global__ void cvt_f32_bf16_kernel(const float* __restrict__ src,
                                    bf16* __restrict__ dst, int n) {
    int i = blockIdx.x * blockDim.x + threadIdx.x;
    if (i < n) dst[i] = f2bf(src[i]);
}

// ------------------------------------------- x (NCHW f32) -> xt (NHWC bf16)
// LDS-tiled 32x32 transpose: coalesced along s on load, along c on store.
__global__ void x_to_nhwc_bf16(const float* __restrict__ x,
                               bf16* __restrict__ xt) {
    __shared__ float tile[32][33];
    int bid = blockIdx.x;
    int st = bid & 127;          // 128 s-tiles
    int ct = (bid >> 7) & 15;    // 16 c-tiles
    int n  = bid >> 11;          // 8 images
    int tid = threadIdx.x;       // 256 threads
    int s0 = st * 32, c0 = ct * 32;

    const float* xb = x + (size_t)n * CIN * S1;
    int sl = tid & 31, cl = tid >> 5;            // cl in 0..7
#pragma unroll
    for (int i = 0; i < 4; ++i)
        tile[cl + i * 8][sl] = xb[(size_t)(c0 + cl + i * 8) * S1 + s0 + sl];
    __syncthreads();

    bf16* xtb = xt + (size_t)n * S1 * CIN;
    int cw = tid & 31, sw = tid >> 5;
#pragma unroll
    for (int i = 0; i < 4; ++i) {
        int s = s0 + sw + i * 8;
        xtb[(size_t)s * CIN + c0 + cw] = f2bf(tile[cw][sw + i * 8]);
    }
}

// ----------------------------------------------- GEMM1: Y1 = Wdown * X (+b)
// A = w_down bf16 [128][512] row-major. Bt = xt [32768][512] (row=col-of-B).
// Y1 written NHWC fp32 [32768][128]. One wave per 16x16 tile, K loop of 16 WMMAs.
__global__ void gemm_down_wmma(const bf16* __restrict__ A,
                               const bf16* __restrict__ Bt,
                               const float* __restrict__ bias,
                               float* __restrict__ Y1) {
    int gtid = blockIdx.x * blockDim.x + threadIdx.x;
    int wave = gtid >> 5;
    int lane = gtid & 31;
    int mt = wave & 7;           // 8 M-tiles (128 rows)
    int st = wave >> 3;          // 2048 S-tiles
    int l16 = lane & 15;
    int kb  = (lane < 16) ? 0 : 8;
    int hi8 = (lane < 16) ? 0 : 8;

    const bf16* arow = A  + (size_t)(mt * 16 + l16) * 512;
    const bf16* brow = Bt + (size_t)(st * 16 + l16) * 512;

    v8f acc = {0.f, 0.f, 0.f, 0.f, 0.f, 0.f, 0.f, 0.f};
#pragma unroll 4
    for (int k0 = 0; k0 < 512; k0 += 32) {
        v16bf a = load_frag16x32(arow, k0, kb);
        v16bf b = load_frag16x32(brow, k0, kb);
        acc = __builtin_amdgcn_wmma_f32_16x16x32_bf16(
            false, a, false, b, (short)0, acc, false, false);
    }
    // C/D: lane -> column s, element r -> row m (=mt*16+hi8+r); NHWC store is
    // 8 contiguous floats per lane.
    int m0 = mt * 16 + hi8;
    float* yrow = Y1 + (size_t)(st * 16 + l16) * MIDC + m0;
#pragma unroll
    for (int r = 0; r < 8; ++r) yrow[r] = acc[r] + bias[m0 + r];
}

// ----------------------------------------------- 3x3 conv: Y2[p][36], fp32
__global__ void conv3x3_enc(const float* __restrict__ Y1,   // [32768][128]
                            const float* __restrict__ Wenc, // [36][128][3][3]
                            const float* __restrict__ benc,
                            float* __restrict__ Y2) {        // [32768][36]
    int idx = blockIdx.x * blockDim.x + threadIdx.x;   // 36*32768 threads
    int e = idx / STOT1;
    int p = idx - e * STOT1;
    int n = p >> 12, s = p & (S1 - 1);
    int h = s >> 6, w = s & 63;

    float acc = benc[e];
    const float* we = Wenc + (size_t)e * MIDC * 9;
#pragma unroll
    for (int i = 0; i < 3; ++i) {
        int hh = h + i - 1;
        if ((unsigned)hh >= (unsigned)HH) continue;
#pragma unroll
        for (int j = 0; j < 3; ++j) {
            int ww = w + j - 1;
            if ((unsigned)ww >= (unsigned)WW) continue;
            const float* yrow = Y1 + ((size_t)(n * S1) + hh * WW + ww) * MIDC;
            const float* wp = we + i * 3 + j;
            for (int c = 0; c < MIDC; ++c)
                acc = fmaf(yrow[c], wp[c * 9], acc);
        }
    }
    Y2[(size_t)p * ENCC + e] = acc;
}

// ------------------- softmax over 9 taps, 4 subpixels; in-place repack to
// row layout [u*9+k]. kern[n,k,h,w,u] == softmax_k(Y2[n, k*4+u, h, w]).
__global__ void softmax_taps(float* __restrict__ Y2) {
    int p = blockIdx.x * blockDim.x + threadIdx.x;   // 32768 pixels
    float* row = Y2 + (size_t)p * ENCC;
    float v[ENCC];
#pragma unroll
    for (int i = 0; i < ENCC; ++i) v[i] = row[i];
#pragma unroll
    for (int u = 0; u < 4; ++u) {
        float m = -1e30f;
#pragma unroll
        for (int k = 0; k < 9; ++k) m = fmaxf(m, v[k * 4 + u]);
        float e[9], sum = 0.f;
#pragma unroll
        for (int k = 0; k < 9; ++k) { e[k] = __expf(v[k * 4 + u] - m); sum += e[k]; }
        float inv = 1.f / sum;
#pragma unroll
        for (int k = 0; k < 9; ++k) row[u * 9 + k] = e[k] * inv;
    }
}

// ------------------- CARAFE reassembly: zt[n][(2h+uy)*128 + 2w+ux][c] (bf16)
// One block per source pixel; 128 threads x 4 channels each.
__global__ void reassemble(const bf16*  __restrict__ xt,   // [32768][512]
                           const float* __restrict__ Kern, // [32768][36] (u*9+k)
                           bf16* __restrict__ zt) {         // [131072][512]
    __shared__ float kk[ENCC];
    int p = blockIdx.x;
    int tid = threadIdx.x;
    if (tid < ENCC) kk[tid] = Kern[(size_t)p * ENCC + tid];
    __syncthreads();

    int n = p >> 12, s = p & (S1 - 1);
    int h = s >> 6, w = s & 63;
    int c0 = tid * 4;

    float acc[4][4] = {};     // [u][cc]
#pragma unroll
    for (int i = 0; i < 3; ++i) {
        int hh = h + i - 1;
        bool okh = (unsigned)hh < (unsigned)HH;
#pragma unroll
        for (int j = 0; j < 3; ++j) {
            int ww = w + j - 1;
            float xv[4] = {0.f, 0.f, 0.f, 0.f};
            if (okh && (unsigned)ww < (unsigned)WW) {
                bf16x4 t = *(const bf16x4*)(
                    xt + (((size_t)(n * S1) + hh * WW + ww) * CIN + c0));
#pragma unroll
                for (int cc = 0; cc < 4; ++cc) xv[cc] = bf2f(t[cc]);
            }
            int k = i * 3 + j;
#pragma unroll
            for (int u = 0; u < 4; ++u) {
                float f = kk[u * 9 + k];
#pragma unroll
                for (int cc = 0; cc < 4; ++cc) acc[u][cc] = fmaf(f, xv[cc], acc[u][cc]);
            }
        }
    }
#pragma unroll
    for (int u = 0; u < 4; ++u) {
        int yy = 2 * h + (u >> 1), xx = 2 * w + (u & 1);
        bf16* zr = zt + (((size_t)n * S2 + yy * W2 + xx) * CIN + c0);
        bf16x4 o;
#pragma unroll
        for (int cc = 0; cc < 4; ++cc) o[cc] = f2bf(acc[u][cc]);
        *(bf16x4*)zr = o;
    }
}

// ------------------- GEMM2: out = Wout * Z (+b), the 68.7 GFLOP hot loop.
// Wave computes 64(M)x16(N): B frag loaded once per K-step, reused by 4 WMMAs.
__global__ void gemm_out_wmma(const bf16* __restrict__ A,   // [512][512]
                              const bf16* __restrict__ Bt,  // [131072][512]
                              const float* __restrict__ bias,
                              float* __restrict__ Out) {     // NCHW fp32
    int gtid = blockIdx.x * blockDim.x + threadIdx.x;
    int wave = gtid >> 5;
    int lane = gtid & 31;
    int mg = wave & 7;           // 8 groups of 64 output channels
    int st = wave >> 3;          // 8192 column tiles (global over N images)
    int l16 = lane & 15;
    int kb  = (lane < 16) ? 0 : 8;
    int hi8 = (lane < 16) ? 0 : 8;

    int sg = st * 16 + l16;                       // global column 0..131071
    const bf16* brow  = Bt + (size_t)sg * 512;
    const bf16* arow0 = A  + (size_t)(mg * 64 + l16) * 512;

    v8f acc[4];
#pragma unroll
    for (int mi = 0; mi < 4; ++mi)
        acc[mi] = (v8f){0.f, 0.f, 0.f, 0.f, 0.f, 0.f, 0.f, 0.f};

    for (int k0 = 0; k0 < 512; k0 += 32) {
        v16bf b = load_frag16x32(brow, k0, kb);
#pragma unroll
        for (int mi = 0; mi < 4; ++mi) {
            v16bf a = load_frag16x32(arow0 + (size_t)mi * 16 * 512, k0, kb);
            acc[mi] = __builtin_amdgcn_wmma_f32_16x16x32_bf16(
                false, a, false, b, (short)0, acc[mi], false, false);
        }
    }

    int n = sg >> 14, s2 = sg & (S2 - 1);
#pragma unroll
    for (int mi = 0; mi < 4; ++mi) {
        int m = mg * 64 + mi * 16 + hi8;
#pragma unroll
        for (int r = 0; r < 8; ++r)
            Out[((size_t)(n * CIN + m + r)) * S2 + s2] = acc[mi][r] + bias[m + r];
    }
}

// ---------------------------------------------------------------------------
extern "C" void kernel_launch(void* const* d_in, const int* in_sizes, int n_in,
                              void* d_out, int out_size, void* d_ws, size_t ws_size,
                              hipStream_t stream) {
    const float* x      = (const float*)d_in[0];
    const float* w_down = (const float*)d_in[1];
    const float* b_down = (const float*)d_in[2];
    const float* w_enc  = (const float*)d_in[3];
    const float* b_enc  = (const float*)d_in[4];
    const float* w_out  = (const float*)d_in[5];
    const float* b_out  = (const float*)d_in[6];
    float* out = (float*)d_out;

    char* ws = (char*)d_ws;
    size_t off = 0;
    auto alloc = [&](size_t bytes) -> void* {
        void* p = ws + off;
        off = (off + bytes + 255) & ~(size_t)255;
        return p;
    };
    bf16*  wdownb = (bf16*) alloc((size_t)MIDC * CIN * 2);          // 128 KB
    bf16*  woutb  = (bf16*) alloc((size_t)CIN * CIN * 2);           // 512 KB
    bf16*  xt     = (bf16*) alloc((size_t)STOT1 * CIN * 2);         // 32 MB
    float* y1     = (float*)alloc((size_t)STOT1 * MIDC * 4);        // 16 MB
    float* y2     = (float*)alloc((size_t)STOT1 * ENCC * 4);        // 4.5 MB
    bf16*  zt     = (bf16*) alloc((size_t)STOT2 * CIN * 2);         // 128 MB
    (void)ws_size; (void)in_sizes; (void)n_in; (void)out_size;

    // 1) weight fp32 -> bf16
    cvt_f32_bf16_kernel<<<(MIDC * CIN + 255) / 256, 256, 0, stream>>>(
        w_down, wdownb, MIDC * CIN);
    cvt_f32_bf16_kernel<<<(CIN * CIN + 255) / 256, 256, 0, stream>>>(
        w_out, woutb, CIN * CIN);
    // 2) x NCHW fp32 -> NHWC bf16
    x_to_nhwc_bf16<<<NN * 16 * 128, 256, 0, stream>>>(x, xt);
    // 3) 1x1 down-projection GEMM (WMMA bf16): 16384 waves
    gemm_down_wmma<<<(8 * (STOT1 / 16)) / 8, 256, 0, stream>>>(
        wdownb, xt, b_down, y1);
    // 4) 3x3 kernel-encoder conv (fp32 VALU)
    conv3x3_enc<<<(ENCC * STOT1) / 256, 256, 0, stream>>>(y1, w_enc, b_enc, y2);
    // 5) softmax over taps, in-place repack
    softmax_taps<<<STOT1 / 256, 256, 0, stream>>>(y2);
    // 6) content reassembly -> upsampled NHWC bf16
    reassemble<<<STOT1, 128, 0, stream>>>(xt, y2, zt);
    // 7) 1x1 output GEMM (WMMA bf16, 64x16 per wave): 65536 waves
    gemm_out_wmma<<<(8 * (STOT2 / 16)) / 8, 256, 0, stream>>>(
        woutb, zt, b_out, out);
}
